// Mamba2Multitask_10196252361273
// MI455X (gfx1250) — compile-verified, matrix operand order checked by reading, not run on hardware
//
#include <hip/hip_runtime.h>
#include <hip/hip_bf16.h>

// ---------------- model constants ----------------
#define D_MODEL   256
#define D_STATE   64
#define D_CONV    4
#define HEADDIM   64
#define D_INNER   512           // EXPAND * D_MODEL
#define NHEADS    8             // D_INNER / HEADDIM
#define CONV_DIM  640           // D_INNER + 2*D_STATE
#define D_IN_PROJ 1160          // 2*D_INNER + 2*D_STATE + NHEADS
#define NUM_LAYERS 4
#define EPS 1e-5f

#define BATCH 16
#define SEQL  2048
#define MROWS (BATCH*SEQL)      // 32768

// padded N for in_proj weight (19 groups of 4 tiles of 16 = 1216 >= 1160)
#define NPAD_IN 1216
#define NT_GRP_IN 19

typedef _Float16 half_t;
typedef __attribute__((ext_vector_type(16))) _Float16 v16h;
typedef __attribute__((ext_vector_type(8)))  _Float16 v8h;
typedef __attribute__((ext_vector_type(8)))  float    v8f;

union frag_u { v16h f; v8h h[2]; };

// A-fragment (and B-fragment, with weights stored [N,K]) for
// v_wmma_f32_16x16x32_f16: lane = row (m = lane&15), the lane holds two
// contiguous 8-half chunks at K offsets kh*8 and 16+kh*8 (kh = lane>>4).
__device__ __forceinline__ v16h load_frag(const half_t* rowbase, int kh) {
    frag_u u;
    u.h[0] = *(const v8h*)(rowbase + kh * 8);
    u.h[1] = *(const v8h*)(rowbase + 16 + kh * 8);
    return u.f;
}

// Stage `bytes` (multiple of 16, from a 16B-aligned source) into LDS using the
// gfx1250 async-copy engine. LDS dest address = low 32 bits of the generic
// pointer (ISA 10.2: LDS aperture maps addr[31:0]).  Tracked by ASYNCcnt.
__device__ __forceinline__ void async_stage(const half_t* __restrict__ src,
                                            half_t* lds_dst, int bytes) {
    unsigned lbase = (unsigned)(size_t)(void*)lds_dst;
    int chunks = bytes >> 4;
    for (int c = threadIdx.x; c < chunks; c += 256) {
        const half_t* g = src + c * 8;
        unsigned l = lbase + c * 16;
        asm volatile("global_load_async_to_lds_b128 %0, %1, off"
                     :: "v"(l), "v"(g) : "memory");
    }
    asm volatile("s_wait_asynccnt 0" ::: "memory");
    __syncthreads();
}

__device__ __forceinline__ float silu_f(float x) {
    return x / (1.0f + __expf(-x));
}
__device__ __forceinline__ float softplus_f(float x) {
    return (x > 20.0f) ? x : log1pf(__expf(x));
}

__device__ __forceinline__ float warp_sum(float v) {
    #pragma unroll
    for (int o = 16; o > 0; o >>= 1) v += __shfl_xor(v, o);
    return v;
}
// blockDim.x == 256 (8 waves)
__device__ __forceinline__ float block_sum(float v, float* lds8) {
    int lane = threadIdx.x & 31, w = threadIdx.x >> 5;
    v = warp_sum(v);
    if (lane == 0) lds8[w] = v;
    __syncthreads();
    float r = (threadIdx.x < 8) ? lds8[threadIdx.x] : 0.0f;
    if (w == 0) {
        r = warp_sum(r);
        if (lane == 0) lds8[0] = r;
    }
    __syncthreads();
    float out = lds8[0];
    __syncthreads();
    return out;
}

// ---------------- kernels ----------------

// h = x @ inp_w + inp_b ; also emit f16 copy for WMMA A-operand
__global__ __launch_bounds__(256)
void k_gemm_in(const float* __restrict__ x, const float* __restrict__ w,
               const float* __restrict__ b, float* __restrict__ h,
               half_t* __restrict__ h16) {
    __shared__ float xs[46];
    int row = blockIdx.x;                 // 0..32767
    int c   = threadIdx.x;                // 0..255
    if (threadIdx.x < 46) xs[threadIdx.x] = x[(size_t)row * 46 + threadIdx.x];
    __syncthreads();
    float acc = b[c];
    #pragma unroll 2
    for (int k = 0; k < 46; ++k) acc += xs[k] * w[k * D_MODEL + c];
    size_t idx = (size_t)row * D_MODEL + c;
    h[idx]   = acc;
    h16[idx] = (half_t)acc;
}

// convert + transpose layer weights to f16 [N,K] layouts
__global__ __launch_bounds__(256)
void k_convert_w(const float* __restrict__ inW,   // [256,1160]
                 const float* __restrict__ outW,  // [512,256]
                 half_t* __restrict__ win16,      // [1216,256]
                 half_t* __restrict__ wout16) {   // [256,512]
    int idx = blockIdx.x * 256 + threadIdx.x;
    const int IN_ELEMS = NPAD_IN * D_MODEL;       // 311296
    if (idx < IN_ELEMS) {
        int n = idx / D_MODEL, k = idx % D_MODEL;
        win16[idx] = (n < D_IN_PROJ) ? (half_t)inW[(size_t)k * D_IN_PROJ + n]
                                     : (half_t)0.0f;
    } else {
        int j = idx - IN_ELEMS;                   // < 131072
        int n = j / D_INNER, k = j % D_INNER;
        wout16[j] = (half_t)outW[(size_t)k * D_MODEL + n];
    }
}

// zxbcdt[M,1160] = h16[M,256] @ in_proj_w  (weights pre-transposed [1216,256]).
// One block = 8 waves sharing a 64-column weight panel, async-staged to LDS.
__global__ __launch_bounds__(256)
void k_gemm_inproj(const half_t* __restrict__ A, const half_t* __restrict__ Bt,
                   float* __restrict__ C) {
    const int K = D_MODEL;
    __shared__ half_t bsh[64 * D_MODEL];          // 32 KB weight panel
    int mg = blockIdx.x / NT_GRP_IN;              // 0..255
    int ng = blockIdx.x % NT_GRP_IN;
    // async-copy the 64xK f16 panel for this n-group into LDS
    async_stage(Bt + (size_t)ng * 64 * K, bsh, 64 * K * 2);

    int wid  = threadIdx.x >> 5;
    int lane = threadIdx.x & 31;
    int mt   = mg * 8 + wid;
    int m = lane & 15, kh = lane >> 4;
    const half_t* arow = A + (size_t)(mt * 16 + m) * K;
    const half_t* brow[4];
    #pragma unroll
    for (int j = 0; j < 4; ++j)
        brow[j] = bsh + (size_t)(j * 16 + m) * K;
    v8f acc[4] = {};
    for (int k0 = 0; k0 < K; k0 += 32) {
        __builtin_prefetch(arow + k0 + 64, 0, 1);
        v16h a = load_frag(arow + k0, kh);
        #pragma unroll
        for (int j = 0; j < 4; ++j) {
            v16h b = load_frag(brow[j] + k0, kh);   // ds_load_b128 x2
            acc[j] = __builtin_amdgcn_wmma_f32_16x16x32_f16(
                false, a, false, b, (short)0, acc[j], false, false);
        }
    }
    int rbase = mt * 16 + 8 * (lane >> 4);        // D layout: row = v + 8*(lane>>4)
    int cl    = lane & 15;                        // col = lane&15
    #pragma unroll
    for (int j = 0; j < 4; ++j) {
        int ncol = ng * 64 + j * 16 + cl;
        if (ncol < D_IN_PROJ) {
            #pragma unroll
            for (int v = 0; v < 8; ++v)
                C[(size_t)(rbase + v) * D_IN_PROJ + ncol] = acc[j][v];
        }
    }
}

// depthwise causal conv (width 4) + bias + silu over xBC slice of zxbcdt
__global__ __launch_bounds__(256)
void k_conv(const float* __restrict__ zx, const float* __restrict__ cw,
            const float* __restrict__ cb, float* __restrict__ xact) {
    int idx = blockIdx.x * 256 + threadIdx.x;     // M*640 total
    int c   = idx % CONV_DIM;
    int row = idx / CONV_DIM;
    int t   = row % SEQL;
    int b   = row / SEQL;
    float w0 = cw[c * 4 + 0], w1 = cw[c * 4 + 1],
          w2 = cw[c * 4 + 2], w3 = cw[c * 4 + 3];
    float acc = cb[c];
    const float* col = zx + (size_t)b * SEQL * D_IN_PROJ + (D_INNER + c);
    if (t >= 3)      acc += w0 * col[(size_t)(t - 3) * D_IN_PROJ];
    if (t >= 2)      acc += w1 * col[(size_t)(t - 2) * D_IN_PROJ];
    if (t >= 1)      acc += w2 * col[(size_t)(t - 1) * D_IN_PROJ];
    acc += w3 * col[(size_t)t * D_IN_PROJ];
    xact[idx] = silu_f(acc);
}

// selective-state scan. grid = B*H*4 blocks; each block owns a 16(p) x 64(n)
// slice of the per-(b,h) 64x64 state in registers (4 floats/thread).
__global__ __launch_bounds__(256)
void k_scan(const float* __restrict__ zx, const float* __restrict__ xact,
            const float* __restrict__ dtb, const float* __restrict__ alog,
            float* __restrict__ yb) {
    __shared__ float sB[D_STATE];
    __shared__ float sC[D_STATE];
    __shared__ float sX[16];
    int bid = blockIdx.x;
    int b   = bid >> 5;                 // / (NHEADS*4)
    int rem = bid & 31;
    int h   = rem >> 2;
    int pg  = rem & 3;
    int p0  = pg * 16;
    int tid = threadIdx.x;
    int pl  = tid >> 4;                 // 0..15 local p
    int n0  = (tid & 15) * 4;           // 4 n per thread
    float Ah  = -__expf(alog[h]);
    float dbi = dtb[h];
    float s0 = 0.f, s1 = 0.f, s2 = 0.f, s3 = 0.f;
    for (int t = 0; t < SEQL; ++t) {
        size_t row = (size_t)b * SEQL + t;
        const float* xr = xact + row * CONV_DIM;
        if (tid < 64)        sB[tid]        = xr[D_INNER + tid];
        else if (tid < 128)  sC[tid - 64]   = xr[D_INNER + D_STATE + (tid - 64)];
        else if (tid < 144)  sX[tid - 128]  = xr[h * HEADDIM + p0 + (tid - 128)];
        __syncthreads();
        float dtv = softplus_f(zx[row * D_IN_PROJ + (D_IN_PROJ - NHEADS) + h] + dbi);
        float dA  = __expf(dtv * Ah);
        float dx  = dtv * sX[pl];
        float b0 = sB[n0], b1 = sB[n0 + 1], b2 = sB[n0 + 2], b3 = sB[n0 + 3];
        float c0 = sC[n0], c1 = sC[n0 + 1], c2 = sC[n0 + 2], c3 = sC[n0 + 3];
        s0 = s0 * dA + dx * b0;
        s1 = s1 * dA + dx * b1;
        s2 = s2 * dA + dx * b2;
        s3 = s3 * dA + dx * b3;
        float acc = s0 * c0 + s1 * c1 + s2 * c2 + s3 * c3;
        // reduce over 16 n-chunk lanes (consecutive within a wave32)
        acc += __shfl_xor(acc, 1);
        acc += __shfl_xor(acc, 2);
        acc += __shfl_xor(acc, 4);
        acc += __shfl_xor(acc, 8);
        if ((tid & 15) == 0)
            yb[row * D_INNER + h * HEADDIM + p0 + pl] = acc;
        __syncthreads();
    }
}

// y = scan_y + D*xh; gate with silu(z); RMS-norm over 512; emit f16
__global__ __launch_bounds__(256)
void k_gate_rms(const float* __restrict__ zx, const float* __restrict__ xact,
                const float* __restrict__ yb, const float* __restrict__ Dp,
                const float* __restrict__ rmsw, half_t* __restrict__ y16) {
    __shared__ float red[8];
    size_t row = blockIdx.x;
    int c0 = threadIdx.x, c1 = threadIdx.x + 256;
    float y0 = yb[row * D_INNER + c0] + Dp[c0 >> 6] * xact[row * CONV_DIM + c0];
    float y1 = yb[row * D_INNER + c1] + Dp[c1 >> 6] * xact[row * CONV_DIM + c1];
    float z0 = zx[row * D_IN_PROJ + c0];
    float z1 = zx[row * D_IN_PROJ + c1];
    float g0 = y0 * silu_f(z0);
    float g1 = y1 * silu_f(z1);
    float tot = block_sum(g0 * g0 + g1 * g1, red);
    float r = rsqrtf(tot * (1.0f / D_INNER) + EPS);
    y16[row * D_INNER + c0] = (half_t)(g0 * r * rmsw[c0]);
    y16[row * D_INNER + c1] = (half_t)(g1 * r * rmsw[c1]);
}

// h[M,256] += y16[M,512] @ out_proj_w (weights pre-transposed [256,512]);
// weight panel async-staged to LDS; also refresh h16 for next layer.
__global__ __launch_bounds__(256)
void k_gemm_outproj(const half_t* __restrict__ A, const half_t* __restrict__ Bt,
                    float* __restrict__ H, half_t* __restrict__ H16) {
    const int K = D_INNER;
    __shared__ half_t bsh[64 * D_INNER];          // 64 KB weight panel
    int mg = blockIdx.x >> 2;                     // 0..255
    int ng = blockIdx.x & 3;
    async_stage(Bt + (size_t)ng * 64 * K, bsh, 64 * K * 2);

    int wid  = threadIdx.x >> 5;
    int lane = threadIdx.x & 31;
    int mt   = mg * 8 + wid;
    int m = lane & 15, kh = lane >> 4;
    const half_t* arow = A + (size_t)(mt * 16 + m) * K;
    const half_t* brow[4];
    #pragma unroll
    for (int j = 0; j < 4; ++j)
        brow[j] = bsh + (size_t)(j * 16 + m) * K;
    v8f acc[4] = {};
    for (int k0 = 0; k0 < K; k0 += 32) {
        __builtin_prefetch(arow + k0 + 64, 0, 1);
        v16h a = load_frag(arow + k0, kh);
        #pragma unroll
        for (int j = 0; j < 4; ++j) {
            v16h b = load_frag(brow[j] + k0, kh);   // ds_load_b128 x2
            acc[j] = __builtin_amdgcn_wmma_f32_16x16x32_f16(
                false, a, false, b, (short)0, acc[j], false, false);
        }
    }
    int rbase = mt * 16 + 8 * (lane >> 4);
    int cl    = lane & 15;
    #pragma unroll
    for (int j = 0; j < 4; ++j) {
        int ncol = ng * 64 + j * 16 + cl;
        #pragma unroll
        for (int v = 0; v < 8; ++v) {
            size_t idx = (size_t)(rbase + v) * D_MODEL + ncol;
            float r = H[idx] + acc[j][v];
            H[idx]   = r;
            H16[idx] = (half_t)r;
        }
    }
}

__global__ __launch_bounds__(256)
void k_zero(float* __restrict__ p, int n) {
    int i = blockIdx.x * 256 + threadIdx.x;
    if (i < n) p[i] = 0.0f;
}

// LayerNorm per (b,l) row + mean over L accumulated into hacc[b,256]
__global__ __launch_bounds__(256)
void k_ln_accum(const float* __restrict__ h, const float* __restrict__ lnw,
                const float* __restrict__ lnb, float* __restrict__ hacc) {
    __shared__ float red[8];
    int b  = blockIdx.x >> 5;           // / 32
    int r0 = (blockIdx.x & 31) * 64;
    int c  = threadIdx.x;
    float w = lnw[c], bb = lnb[c];
    float accv = 0.0f;
    for (int r = 0; r < 64; ++r) {
        size_t row = (size_t)b * SEQL + r0 + r;
        float v  = h[row * D_MODEL + c];
        float s  = block_sum(v, red);
        float sq = block_sum(v * v, red);
        float mu  = s * (1.0f / D_MODEL);
        float var = sq * (1.0f / D_MODEL) - mu * mu;
        accv += (v - mu) * rsqrtf(var + EPS) * w + bb;
    }
    atomicAdd(&hacc[b * D_MODEL + c], accv * (1.0f / SEQL));
}

// final heads: d_out[0..15] = dir, d_out[16..63] = reg (b-major)
__global__ __launch_bounds__(64)
void k_head(const float* __restrict__ hacc, const float* __restrict__ dw,
            const float* __restrict__ db, const float* __restrict__ rw,
            const float* __restrict__ rb, float* __restrict__ out) {
    int i = threadIdx.x;
    if (i < 16) {
        float s = db[0];
        for (int k = 0; k < D_MODEL; ++k) s += hacc[i * D_MODEL + k] * dw[k];
        out[i] = s;
    } else if (i < 64) {
        int j = i - 16, b = j / 3, cc = j % 3;
        float s = rb[cc];
        for (int k = 0; k < D_MODEL; ++k) s += hacc[b * D_MODEL + k] * rw[k * 3 + cc];
        out[16 + j] = s;
    }
}

// ---------------- host side ----------------
static inline size_t align256(size_t x) { return (x + 255) & ~(size_t)255; }

extern "C" void kernel_launch(void* const* d_in, const int* in_sizes, int n_in,
                              void* d_out, int out_size, void* d_ws, size_t ws_size,
                              hipStream_t stream) {
    (void)in_sizes; (void)n_in; (void)out_size; (void)ws_size;
    const float* x        = (const float*)d_in[0];
    const float* inp_w    = (const float*)d_in[1];
    const float* inp_b    = (const float*)d_in[2];
    const float* in_projw = (const float*)d_in[3];   // [4,256,1160]
    const float* conv_w   = (const float*)d_in[4];   // [4,640,4]
    const float* conv_b   = (const float*)d_in[5];   // [4,640]
    const float* dt_bias  = (const float*)d_in[6];   // [4,8]
    const float* a_log    = (const float*)d_in[7];   // [4,8]
    const float* Dp       = (const float*)d_in[8];   // [4,8]
    const float* rms_w    = (const float*)d_in[9];   // [4,512]
    const float* out_projw= (const float*)d_in[10];  // [4,512,256]
    const float* ln_w     = (const float*)d_in[11];
    const float* ln_b     = (const float*)d_in[12];
    const float* dir_w    = (const float*)d_in[13];
    const float* dir_b    = (const float*)d_in[14];
    const float* reg_w    = (const float*)d_in[15];
    const float* reg_b    = (const float*)d_in[16];

    char* ws = (char*)d_ws;
    size_t off = 0;
    float*  h     = (float*)(ws + off); off = align256(off + (size_t)MROWS * D_MODEL * 4);
    float*  zx    = (float*)(ws + off); off = align256(off + (size_t)MROWS * D_IN_PROJ * 4);
    float*  xact  = (float*)(ws + off); off = align256(off + (size_t)MROWS * CONV_DIM * 4);
    float*  yb    = (float*)(ws + off); off = align256(off + (size_t)MROWS * D_INNER * 4);
    half_t* h16   = (half_t*)(ws + off); off = align256(off + (size_t)MROWS * D_MODEL * 2);
    half_t* y16   = (half_t*)(ws + off); off = align256(off + (size_t)MROWS * D_INNER * 2);
    half_t* win16 = (half_t*)(ws + off); off = align256(off + (size_t)NPAD_IN * D_MODEL * 2);
    half_t* wout16= (half_t*)(ws + off); off = align256(off + (size_t)D_MODEL * D_INNER * 2);
    float*  hacc  = (float*)(ws + off); off = align256(off + (size_t)BATCH * D_MODEL * 4);

    // stem: h = x @ inp_w + inp_b (also f16 copy)
    k_gemm_in<<<MROWS, 256, 0, stream>>>(x, inp_w, inp_b, h, h16);

    for (int i = 0; i < NUM_LAYERS; ++i) {
        k_convert_w<<<1728, 256, 0, stream>>>(
            in_projw + (size_t)i * D_MODEL * D_IN_PROJ,
            out_projw + (size_t)i * D_INNER * D_MODEL, win16, wout16);
        k_gemm_inproj<<<(MROWS / 16 / 8) * NT_GRP_IN, 256, 0, stream>>>(h16, win16, zx);
        k_conv<<<(MROWS * CONV_DIM) / 256, 256, 0, stream>>>(
            zx, conv_w + (size_t)i * CONV_DIM * D_CONV, conv_b + (size_t)i * CONV_DIM, xact);
        k_scan<<<BATCH * NHEADS * 4, 256, 0, stream>>>(
            zx, xact, dt_bias + i * NHEADS, a_log + i * NHEADS, yb);
        k_gate_rms<<<MROWS, 256, 0, stream>>>(
            zx, xact, yb, Dp + i * NHEADS, rms_w + (size_t)i * D_INNER, y16);
        k_gemm_outproj<<<(MROWS / 16 / 8) * 4, 256, 0, stream>>>(y16, wout16, h, h16);
    }

    k_zero<<<(BATCH * D_MODEL) / 256, 256, 0, stream>>>(hacc, BATCH * D_MODEL);
    k_ln_accum<<<BATCH * 32, 256, 0, stream>>>(h, ln_w, ln_b, hacc);
    k_head<<<1, 64, 0, stream>>>(hacc, dir_w, dir_b, reg_w, reg_b, (float*)d_out);
}
